// WaveBlock_67508295958996
// MI455X (gfx1250) — matrix operand order repaired and checked by reading, not compiled, not run on hardware
//
#include <hip/hip_runtime.h>
#include <math.h>

// ---------------------------------------------------------------------------
// WaveBlock for MI455X (gfx1250). All GEMMs lower to v_wmma_f32_16x16x32_f16.
// Shapes: B=2, N=1024, D=512, H=8, DH=64, HF=2048, R=16. All GEMM dims are
// multiples of the 128x64x32 block tile, so no edge guards are needed.
// expm is computed by Taylor series (||A|| ~ 0.02 -> 8 terms is converged),
// turning the matrix exponential into a chain of 512^3 WMMA GEMMs.
//
// Complex-multiply signs are folded into the LDS staging (negated Ai copy and
// conj-signed Bi copy), so the WMMA inner loop is pure wmma-accumulate with no
// per-fragment VALU sign fixups. The GEMM is templated on the three variants
// actually used so no runtime flag selects survive in the hot loop.
// ---------------------------------------------------------------------------

#define B_   2
#define N_   1024
#define D_   512
#define H_   8
#define DH_  64
#define HF_  2048
#define R_   16
#define M2_  (B_ * N_)   // 2048 tokens
#define BH_  (B_ * H_)   // 16 (batch*heads)

typedef _Float16 h8   __attribute__((ext_vector_type(8)));
typedef _Float16 v16h __attribute__((ext_vector_type(16)));
typedef float    v8f  __attribute__((ext_vector_type(8)));

__device__ __forceinline__ v8f wmma_f16(v16h a, v16h b, v8f c) {
  // (neg_a, A, neg_b, B, c_mod, C, reuse_a, reuse_b)
  return __builtin_amdgcn_wmma_f32_16x16x32_f16(false, a, false, b, (short)0, c,
                                                false, false);
}

// ---------------------------------------------------------------------------
// Batched complex GEMM:  C = alpha * A x B(')  (+ bias),  f16 WMMA, f32 accum
//   BT=1 : B given as [N][K] row-major (weights / "B^T" access)
//   BT=0 : B given as [K][N] row-major
//   CONJB: use conj(B) (for q k^H attention scores); sign folded into staging
//   HAS_AI / HAS_BI: compile-time real/complex selection
// Block tile 128(M) x 64(N), 8 waves as 4x2 grid, each wave 32x32 (2x2 WMMA
// tiles), K staged in LDS in 32-wide steps as f16 with +8 half padding.
//   Cr = Ar*Br + (-Ai)*Bi'   Ci = Ai*Br + Ar*Bi'   with Bi' = (CONJB? -Bi:Bi)
// ---------------------------------------------------------------------------
#define TM 128
#define TN 64
#define TK 32
#define LDS_K 40   // 32 + 8 pad halves (80B row stride, 16B aligned)

template <bool CONJB, bool HAS_AI, bool HAS_BI, bool BT>
__global__ __launch_bounds__(256) void k_gemm_cplx(
    const float* __restrict__ Ar, const float* __restrict__ Ai, int lda, long long sA,
    const float* __restrict__ Br, const float* __restrict__ Bi, int ldb, long long sB,
    float* __restrict__ Cr, float* __restrict__ Ci, int ldc, long long sC,
    int K, float alpha,
    const float* __restrict__ biasR, const float* __restrict__ biasI)
{
  __shared__ _Float16 As_r[TM][LDS_K];
  __shared__ _Float16 As_i[TM][LDS_K];    // +Ai   (used when HAS_AI)
  __shared__ _Float16 As_n[TM][LDS_K];    // -Ai   (used when HAS_AI && HAS_BI)
  __shared__ _Float16 Bs_r[TN][LDS_K];
  __shared__ _Float16 Bs_i[TN][LDS_K];    // conj-signed Bi (used when HAS_BI)

  const int t    = threadIdx.x;
  const int wave = t >> 5;
  const int lane = t & 31;
  const int z    = blockIdx.z;
  const long long m0 = (long long)blockIdx.y * TM;
  const long long n0 = (long long)blockIdx.x * TN;

  Ar += (long long)z * sA;
  if (HAS_AI) Ai += (long long)z * sA;
  Br += (long long)z * sB;
  if (HAS_BI) Bi += (long long)z * sB;
  Cr += (long long)z * sC;  Ci += (long long)z * sC;

  const int waveM = wave >> 1;  // 0..3 -> 32-row slab
  const int waveN = wave & 1;   // 0..1 -> 32-col slab

  v8f accR[2][2] = {};
  v8f accI[2][2] = {};

  const int rA  = t >> 3;        // 0..31
  const int cA  = (t & 7) * 4;   // 0,4,...,28
  const int nB  = t & 63;
  const int kB0 = t >> 6;        // 0..3
  const float bsign = CONJB ? -1.0f : 1.0f;

  for (int kk = 0; kk < K; kk += TK) {
    // ---- prefetch next K tile (global_prefetch_b8) ----
    if (kk + TK < K) {
      __builtin_prefetch(Ar + (m0 + rA) * lda + kk + TK + cA, 0, 1);
      if (BT) __builtin_prefetch(Br + (n0 + (rA & 31)) * ldb + kk + TK + cA, 0, 1);
    }
    // ---- stage A tile (128 x 32) as f16 ----
#pragma unroll
    for (int it = 0; it < 4; ++it) {
      const int m = it * 32 + rA;
      const float4 fr = *(const float4*)(Ar + (m0 + m) * lda + kk + cA);
      As_r[m][cA + 0] = (_Float16)fr.x;  As_r[m][cA + 1] = (_Float16)fr.y;
      As_r[m][cA + 2] = (_Float16)fr.z;  As_r[m][cA + 3] = (_Float16)fr.w;
      if (HAS_AI) {
        const float4 fi = *(const float4*)(Ai + (m0 + m) * lda + kk + cA);
        As_i[m][cA + 0] = (_Float16)fi.x;  As_i[m][cA + 1] = (_Float16)fi.y;
        As_i[m][cA + 2] = (_Float16)fi.z;  As_i[m][cA + 3] = (_Float16)fi.w;
        if (HAS_BI) {
          As_n[m][cA + 0] = (_Float16)(-fi.x);  As_n[m][cA + 1] = (_Float16)(-fi.y);
          As_n[m][cA + 2] = (_Float16)(-fi.z);  As_n[m][cA + 3] = (_Float16)(-fi.w);
        }
      }
    }
    // ---- stage B tile (64 x 32), stored [n][k] in LDS ----
    if (BT) {
#pragma unroll
      for (int it = 0; it < 2; ++it) {
        const int n = it * 32 + rA;
        const float4 fr = *(const float4*)(Br + (n0 + n) * ldb + kk + cA);
        Bs_r[n][cA + 0] = (_Float16)fr.x;  Bs_r[n][cA + 1] = (_Float16)fr.y;
        Bs_r[n][cA + 2] = (_Float16)fr.z;  Bs_r[n][cA + 3] = (_Float16)fr.w;
        if (HAS_BI) {
          const float4 fi = *(const float4*)(Bi + (n0 + n) * ldb + kk + cA);
          Bs_i[n][cA + 0] = (_Float16)(bsign * fi.x);
          Bs_i[n][cA + 1] = (_Float16)(bsign * fi.y);
          Bs_i[n][cA + 2] = (_Float16)(bsign * fi.z);
          Bs_i[n][cA + 3] = (_Float16)(bsign * fi.w);
        }
      }
    } else {
#pragma unroll
      for (int it = 0; it < 8; ++it) {
        const int kl = kB0 + it * 4;
        Bs_r[nB][kl] = (_Float16)Br[(long long)(kk + kl) * ldb + n0 + nB];
        if (HAS_BI)
          Bs_i[nB][kl] = (_Float16)(bsign * Bi[(long long)(kk + kl) * ldb + n0 + nB]);
      }
    }
    __syncthreads();

    // ---- gather fragments (ISA 16-bit A/B VGPR layouts) & WMMA ----
    {
      const int l15 = lane & 15;
      const int hiA = (lane >> 4) * 8;    // A: K = hi*8 + e (e<8), 16+hi*8+e-8
      const int hiB = (lane >> 4) * 8;    // B: K = hi*8 + e (e<8), 16+hi*8+e-8
      v16h arf[2], aif[2], anf[2], brf[2], bif[2];
#pragma unroll
      for (int mt = 0; mt < 2; ++mt) {
        const int row = waveM * 32 + mt * 16 + l15;
        h8 lo = *(const h8*)&As_r[row][hiA];
        h8 hi = *(const h8*)&As_r[row][16 + hiA];
        arf[mt] = __builtin_shufflevector(lo, hi, 0,1,2,3,4,5,6,7,
                                                  8,9,10,11,12,13,14,15);
        if (HAS_AI) {
          h8 lo2 = *(const h8*)&As_i[row][hiA];
          h8 hi2 = *(const h8*)&As_i[row][16 + hiA];
          aif[mt] = __builtin_shufflevector(lo2, hi2, 0,1,2,3,4,5,6,7,
                                                      8,9,10,11,12,13,14,15);
          if (HAS_BI) {
            h8 lo3 = *(const h8*)&As_n[row][hiA];
            h8 hi3 = *(const h8*)&As_n[row][16 + hiA];
            anf[mt] = __builtin_shufflevector(lo3, hi3, 0,1,2,3,4,5,6,7,
                                                        8,9,10,11,12,13,14,15);
          }
        }
      }
#pragma unroll
      for (int nt = 0; nt < 2; ++nt) {
        const int col = waveN * 32 + nt * 16 + l15;
        h8 lo = *(const h8*)&Bs_r[col][hiB];
        h8 hi = *(const h8*)&Bs_r[col][16 + hiB];
        brf[nt] = __builtin_shufflevector(lo, hi, 0,1,2,3,4,5,6,7,
                                                  8,9,10,11,12,13,14,15);
        if (HAS_BI) {
          h8 lo2 = *(const h8*)&Bs_i[col][hiB];
          h8 hi2 = *(const h8*)&Bs_i[col][16 + hiB];
          bif[nt] = __builtin_shufflevector(lo2, hi2, 0,1,2,3,4,5,6,7,
                                                      8,9,10,11,12,13,14,15);
        }
      }
#pragma unroll
      for (int mt = 0; mt < 2; ++mt) {
#pragma unroll
        for (int nt = 0; nt < 2; ++nt) {
          accR[mt][nt] = wmma_f16(arf[mt], brf[nt], accR[mt][nt]);
          if (HAS_AI) accI[mt][nt] = wmma_f16(aif[mt], brf[nt], accI[mt][nt]);
          if (HAS_BI) {
            if (HAS_AI)
              accR[mt][nt] = wmma_f16(anf[mt], bif[nt], accR[mt][nt]);
            accI[mt][nt] = wmma_f16(arf[mt], bif[nt], accI[mt][nt]);
          }
        }
      }
    }
    __syncthreads();
  }

  // ---- epilogue: 32-bit C/D layout (lanes 0-15: M=v, lanes 16-31: M=8+v) ----
  const int l15 = lane & 15;
  const int hiC = lane >> 4;
#pragma unroll
  for (int mt = 0; mt < 2; ++mt) {
#pragma unroll
    for (int nt = 0; nt < 2; ++nt) {
      const long long col = n0 + waveN * 32 + nt * 16 + l15;
      const float brv = biasR ? biasR[col] : 0.0f;
      const float biv = biasI ? biasI[col] : 0.0f;
#pragma unroll
      for (int v = 0; v < 8; ++v) {
        const long long row = m0 + waveM * 32 + mt * 16 + hiC * 8 + v;
        Cr[row * ldc + col] = alpha * accR[mt][nt][v] + brv;
        Ci[row * ldc + col] = alpha * accI[mt][nt][v] + biv;
      }
    }
  }
}

// ---------------------------------------------------------------------------
// Small kernels
// ---------------------------------------------------------------------------

// router input: [B,4D] = concat(interleave(mean_r,mean_i), interleave(psi))
__global__ __launch_bounds__(256) void k_mean_rin(
    const float* __restrict__ xr, const float* __restrict__ xi,
    const float* __restrict__ pr, const float* __restrict__ pi,
    float* __restrict__ rin)
{
  const int idx = blockIdx.x * 256 + threadIdx.x;
  if (idx >= B_ * D_) return;
  const int b = idx / D_, d = idx % D_;
  const float* br = xr + (long long)b * N_ * D_ + d;
  const float* bi = xi + (long long)b * N_ * D_ + d;
  float sr = 0.f, si = 0.f;
  for (int n = 0; n < N_; ++n) { sr += br[(long long)n * D_]; si += bi[(long long)n * D_]; }
  sr *= (1.0f / N_);  si *= (1.0f / N_);
  float* o = rin + (long long)b * 4 * D_;
  o[2 * d] = sr;               o[2 * d + 1] = si;
  o[2 * D_ + 2 * d] = pr[b * D_ + d];  o[2 * D_ + 2 * d + 1] = pi[b * D_ + d];
}

__global__ __launch_bounds__(256) void k_router_hid(
    const float* __restrict__ rin, const float* __restrict__ W1,
    const float* __restrict__ b1, float* __restrict__ hid)
{
  const int idx = blockIdx.x * 256 + threadIdx.x;
  if (idx >= B_ * D_) return;
  const int b = idx / D_, e = idx % D_;
  const float* r = rin + (long long)b * 4 * D_;
  const float* w = W1 + (long long)e * 4 * D_;
  float s = b1[e];
  for (int j = 0; j < 4 * D_; ++j) s += r[j] * w[j];
  hid[idx] = fmaxf(s, 0.0f);
}

__global__ __launch_bounds__(256) void k_router_r(
    const float* __restrict__ hid, const float* __restrict__ W2,
    const float* __restrict__ b2, float* __restrict__ rout)
{
  __shared__ float red[256];
  const int b = blockIdx.x, t = threadIdx.x;
  float s = 0.f;
  for (int e = t; e < D_; e += 256) s += hid[b * D_ + e] * W2[e];
  red[t] = s;  __syncthreads();
  for (int st = 128; st > 0; st >>= 1) {
    if (t < st) red[t] += red[t + st];
    __syncthreads();
  }
  if (t == 0) rout[b] = 1.0f / (1.0f + expf(-(red[0] + b2[0])));
}

// split [2048, 3*512] qkv GEMM output into planar [BH][N][DH] q/k/v
__global__ __launch_bounds__(256) void k_split_qkv(
    const float* __restrict__ cr, const float* __restrict__ ci,
    float* __restrict__ qr, float* __restrict__ qi,
    float* __restrict__ kr, float* __restrict__ ki,
    float* __restrict__ vr, float* __restrict__ vi)
{
  const long long idx = (long long)blockIdx.x * 256 + threadIdx.x;
  const long long per = (long long)M2_ * D_;
  if (idx >= 3 * per) return;
  const int which = (int)(idx / per);
  const long long rem = idx - (long long)which * per;
  const int m = (int)(rem / D_), e = (int)(rem % D_);
  const int b = m / N_, n = m % N_;
  const int h = e >> 6, dh = e & 63;
  const long long src = (long long)m * (3 * D_) + (long long)which * D_ + e;
  const long long dst = ((long long)(b * H_ + h) * N_ + n) * DH_ + dh;
  float* pr = which == 0 ? qr : (which == 1 ? kr : vr);
  float* pi = which == 0 ? qi : (which == 1 ? ki : vi);
  pr[dst] = cr[src];  pi[dst] = ci[src];
}

// planar [BH][N][DH] -> [2048, 512]
__global__ __launch_bounds__(256) void k_merge_heads(
    const float* __restrict__ ar, const float* __restrict__ ai,
    float* __restrict__ mr, float* __restrict__ mi)
{
  const long long idx = (long long)blockIdx.x * 256 + threadIdx.x;
  if (idx >= (long long)M2_ * D_) return;
  const int m = (int)(idx / D_), e = (int)(idx % D_);
  const int b = m / N_, n = m % N_;
  const int h = e >> 6, dh = e & 63;
  const long long src = ((long long)(b * H_ + h) * N_ + n) * DH_ + dh;
  mr[idx] = ar[src];  mi[idx] = ai[src];
}

// in-place 1024-point radix-2 complex FFT over the N axis of [BH][N][DH],
// ortho norm (1/sqrt(N) each direction). dirsign = -1 fwd, +1 inv.
__global__ __launch_bounds__(256) void k_fft1024(
    float* __restrict__ re, float* __restrict__ im, float dirsign)
{
  __shared__ float sr[N_];
  __shared__ float si[N_];
  const int c = blockIdx.x;             // 0 .. BH*DH-1
  const int bh = c >> 6, d = c & 63;
  const long long base = (long long)bh * N_ * DH_ + d;
  const int t = threadIdx.x;
  for (int j = t; j < N_; j += 256) {
    const unsigned rj = __brev((unsigned)j) >> 22;   // 10-bit reverse
    sr[rj] = re[base + (long long)j * DH_];
    si[rj] = im[base + (long long)j * DH_];
  }
  __syncthreads();
  for (int len = 2; len <= N_; len <<= 1) {
    const int half = len >> 1;
    const float ang = dirsign * 6.28318530717958647f / (float)len;
    for (int u = t; u < N_ / 2; u += 256) {
      const int blk = u / half, pos = u - blk * half;
      const int i0 = blk * len + pos, i1 = i0 + half;
      float ws, wc;
      __sincosf(ang * (float)pos, &ws, &wc);
      const float xr = sr[i1], xi = si[i1];
      const float tr = wc * xr - ws * xi;
      const float ti = wc * xi + ws * xr;
      const float ur = sr[i0], ui = si[i0];
      sr[i0] = ur + tr;  si[i0] = ui + ti;
      sr[i1] = ur - tr;  si[i1] = ui - ti;
    }
    __syncthreads();
  }
  const float sc = 0.03125f;  // 1/sqrt(1024)
  for (int j = t; j < N_; j += 256) {
    re[base + (long long)j * DH_] = sr[j] * sc;
    im[base + (long long)j * DH_] = si[j] * sc;
  }
}

// attn = softmax(|sim|) written in-place over sim_r (one block per row)
__global__ __launch_bounds__(256) void k_softmax_abs(
    float* __restrict__ sr, const float* __restrict__ si)
{
  __shared__ float red[256];
  const long long base = (long long)blockIdx.x * N_;
  const int t = threadIdx.x;
  float mx = -1e30f;
  for (int j = t; j < N_; j += 256) {
    const float a = sr[base + j], b = si[base + j];
    mx = fmaxf(mx, sqrtf(a * a + b * b));
  }
  red[t] = mx;  __syncthreads();
  for (int s = 128; s > 0; s >>= 1) {
    if (t < s) red[t] = fmaxf(red[t], red[t + s]);
    __syncthreads();
  }
  mx = red[0];  __syncthreads();
  float sum = 0.f;
  for (int j = t; j < N_; j += 256) {
    const float a = sr[base + j], b = si[base + j];
    const float e = __expf(sqrtf(a * a + b * b) - mx);
    sr[base + j] = e;  sum += e;
  }
  red[t] = sum;  __syncthreads();
  for (int s = 128; s > 0; s >>= 1) {
    if (t < s) red[t] += red[t + s];
    __syncthreads();
  }
  const float inv = 1.0f / red[0];
  for (int j = t; j < N_; j += 256) sr[base + j] *= inv;
}

__global__ __launch_bounds__(256) void k_combine(
    const float* __restrict__ xr, const float* __restrict__ xi,
    const float* __restrict__ aqr, const float* __restrict__ aqi,
    const float* __restrict__ afr, const float* __restrict__ afi,
    const float* __restrict__ rbuf, float* __restrict__ outr, float* __restrict__ outi)
{
  const long long idx = (long long)blockIdx.x * 256 + threadIdx.x;
  if (idx >= (long long)M2_ * D_) return;
  const int b = (int)(idx / ((long long)N_ * D_));
  const float rv = rbuf[b];
  outr[idx] = xr[idx] + rv * aqr[idx] + (1.0f - rv) * afr[idx];
  outi[idx] = xi[idx] + rv * aqi[idx] + (1.0f - rv) * afi[idx];
}

// LayerNorm over interleaved real view (2D values per token), in place
__global__ __launch_bounds__(256) void k_layernorm(
    float* __restrict__ xr, float* __restrict__ xi,
    const float* __restrict__ g, const float* __restrict__ bb)
{
  __shared__ float s1[256];
  __shared__ float s2[256];
  const long long base = (long long)blockIdx.x * D_;
  const int t = threadIdx.x;
  float su = 0.f, sq = 0.f;
  for (int j = t; j < D_; j += 256) {
    const float a = xr[base + j], c = xi[base + j];
    su += a + c;  sq += a * a + c * c;
  }
  s1[t] = su;  s2[t] = sq;  __syncthreads();
  for (int s = 128; s > 0; s >>= 1) {
    if (t < s) { s1[t] += s1[t + s];  s2[t] += s2[t + s]; }
    __syncthreads();
  }
  const float mean = s1[0] * (1.0f / (2.0f * D_));
  const float var  = s2[0] * (1.0f / (2.0f * D_)) - mean * mean;
  const float rstd = rsqrtf(var + 1e-5f);
  for (int j = t; j < D_; j += 256) {
    const float a = xr[base + j], c = xi[base + j];
    xr[base + j] = (a - mean) * rstd * g[2 * j]     + bb[2 * j];
    xi[base + j] = (c - mean) * rstd * g[2 * j + 1] + bb[2 * j + 1];
  }
}

__global__ __launch_bounds__(256) void k_cgelu(
    float* __restrict__ hr, float* __restrict__ hi, long long total)
{
  const long long idx = (long long)blockIdx.x * 256 + threadIdx.x;
  if (idx >= total) return;
  const float a = hr[idx], b = hi[idx];
  hr[idx] = 0.5f * a * (1.0f + erff(a * 0.70710678118654752f));
  hi[idx] = 0.5f * b * (1.0f + erff(b * 0.70710678118654752f));
}

__global__ __launch_bounds__(256) void k_add2(
    float* __restrict__ ar, float* __restrict__ ai,
    const float* __restrict__ br, const float* __restrict__ bi, long long total)
{
  const long long idx = (long long)blockIdx.x * 256 + threadIdx.x;
  if (idx >= total) return;
  ar[idx] += br[idx];  ai[idx] += bi[idx];
}

// H = diag(eps) + V V^H ; A = -i*dt*H (Ar = dt*Hi, Ai = -dt*Hr); also write A^T
__global__ __launch_bounds__(256) void k_hbuild(
    const float* __restrict__ epsr, const float* __restrict__ epsi,
    const float* __restrict__ Vr, const float* __restrict__ Vi,
    const float* __restrict__ dtp,
    float* __restrict__ Ar, float* __restrict__ Ai,
    float* __restrict__ ATr, float* __restrict__ ATi)
{
  const int idx = blockIdx.x * 256 + threadIdx.x;
  if (idx >= D_ * D_) return;
  const int i = idx / D_, j = idx % D_;
  float hrv = 0.f, hiv = 0.f;
  for (int r = 0; r < R_; ++r) {
    const float vir = Vr[i * R_ + r], vii = Vi[i * R_ + r];
    const float vjr = Vr[j * R_ + r], vji = Vi[j * R_ + r];
    hrv += vir * vjr + vii * vji;          // v_i * conj(v_j)
    hiv += vii * vjr - vir * vji;
  }
  if (i == j) { hrv += epsr[i];  hiv += epsi[i]; }
  const float dt = dtp[0];
  const float ar = dt * hiv;
  const float ai = -dt * hrv;
  Ar[idx] = ar;  Ai[idx] = ai;
  ATr[j * D_ + i] = ar;  ATi[j * D_ + i] = ai;
}

__global__ __launch_bounds__(256) void k_uinit(
    const float* __restrict__ Ar, const float* __restrict__ Ai,
    float* __restrict__ Ur, float* __restrict__ Ui)
{
  const int idx = blockIdx.x * 256 + threadIdx.x;
  if (idx >= D_ * D_) return;
  const int i = idx / D_, j = idx % D_;
  Ur[idx] = Ar[idx] + (i == j ? 1.0f : 0.0f);
  Ui[idx] = Ai[idx];
}

// pack complex result as interleaved (re,im) pairs, then r[b]
__global__ __launch_bounds__(256) void k_pack(
    const float* __restrict__ yr, const float* __restrict__ yi,
    const float* __restrict__ rbuf, float* __restrict__ out,
    long long nelem, long long out_size)
{
  const long long idx = (long long)blockIdx.x * 256 + threadIdx.x;
  if (idx < nelem) {
    if (2 * idx + 1 < out_size) {
      out[2 * idx]     = yr[idx];
      out[2 * idx + 1] = yi[idx];
    }
  }
  if (idx < B_ && 2 * nelem + idx < out_size) out[2 * nelem + idx] = rbuf[idx];
}

// ---------------------------------------------------------------------------
// Host orchestration
// ---------------------------------------------------------------------------
// variant 0: complex A, complex B[N][K]          (projections, FFN, Taylor)
// variant 1: complex A, conj(complex B)[N][K]    (attention scores)
// variant 2: real A,    complex B[K][N]          (attn @ v)
static inline void gemm_c(hipStream_t s, int variant,
    const float* Ar, const float* Ai, int lda, long long sA,
    const float* Br, const float* Bi, int ldb, long long sB,
    float* Cr, float* Ci, int ldc, long long sC,
    int M, int Nn, int K, int batch, float alpha,
    const float* biasR, const float* biasI)
{
  dim3 g(Nn / TN, M / TM, batch), blk(256, 1, 1);
  if (variant == 0) {
    k_gemm_cplx<false, true, true, true><<<g, blk, 0, s>>>(
        Ar, Ai, lda, sA, Br, Bi, ldb, sB, Cr, Ci, ldc, sC, K, alpha, biasR, biasI);
  } else if (variant == 1) {
    k_gemm_cplx<true, true, true, true><<<g, blk, 0, s>>>(
        Ar, Ai, lda, sA, Br, Bi, ldb, sB, Cr, Ci, ldc, sC, K, alpha, biasR, biasI);
  } else {
    k_gemm_cplx<false, false, true, false><<<g, blk, 0, s>>>(
        Ar, Ai, lda, sA, Br, Bi, ldb, sB, Cr, Ci, ldc, sC, K, alpha, biasR, biasI);
  }
}

extern "C" void kernel_launch(void* const* d_in, const int* in_sizes, int n_in,
                              void* d_out, int out_size, void* d_ws, size_t ws_size,
                              hipStream_t stream) {
  (void)in_sizes; (void)n_in; (void)ws_size;
  const float* x_real  = (const float*)d_in[0];
  const float* x_imag  = (const float*)d_in[1];
  const float* psi_r   = (const float*)d_in[2];
  const float* psi_i   = (const float*)d_in[3];
  const float* qa_qkv_Wr = (const float*)d_in[4];
  const float* qa_qkv_Wi = (const float*)d_in[5];
  const float* qa_qkv_br = (const float*)d_in[6];
  const float* qa_qkv_bi = (const float*)d_in[7];
  const float* qa_o_Wr   = (const float*)d_in[8];
  const float* qa_o_Wi   = (const float*)d_in[9];
  const float* qa_o_br   = (const float*)d_in[10];
  const float* qa_o_bi   = (const float*)d_in[11];
  const float* fa_qkv_Wr = (const float*)d_in[12];
  const float* fa_qkv_Wi = (const float*)d_in[13];
  const float* fa_qkv_br = (const float*)d_in[14];
  const float* fa_qkv_bi = (const float*)d_in[15];
  const float* fa_o_Wr   = (const float*)d_in[16];
  const float* fa_o_Wi   = (const float*)d_in[17];
  const float* fa_o_br   = (const float*)d_in[18];
  const float* fa_o_bi   = (const float*)d_in[19];
  const float* rt_W1     = (const float*)d_in[20];
  const float* rt_b1     = (const float*)d_in[21];
  const float* rt_W2     = (const float*)d_in[22];
  const float* rt_b2     = (const float*)d_in[23];
  const float* ff1_Wr    = (const float*)d_in[24];
  const float* ff1_Wi    = (const float*)d_in[25];
  const float* ff1_br    = (const float*)d_in[26];
  const float* ff1_bi    = (const float*)d_in[27];
  const float* ff2_Wr    = (const float*)d_in[28];
  const float* ff2_Wi    = (const float*)d_in[29];
  const float* ff2_br    = (const float*)d_in[30];
  const float* ff2_bi    = (const float*)d_in[31];
  const float* ev_eps_r  = (const float*)d_in[32];
  const float* ev_eps_i  = (const float*)d_in[33];
  const float* ev_V_r    = (const float*)d_in[34];
  const float* ev_V_i    = (const float*)d_in[35];
  const float* ev_dt     = (const float*)d_in[36];
  const float* ln1_g     = (const float*)d_in[37];
  const float* ln1_b     = (const float*)d_in[38];
  const float* ln2_g     = (const float*)d_in[39];
  const float* ln2_b     = (const float*)d_in[40];

  // workspace bump allocator (floats, 64-element aligned)
  float* wsf = (float*)d_ws;
  size_t off = 0;
  auto alloc = [&](size_t n) -> float* {
    float* p = wsf + off;
    off += (n + 63) & ~((size_t)63);
    return p;
  };
  const long long TOK = (long long)M2_ * D_;   // 1,048,576
  const long long DD  = (long long)D_ * D_;    // 262,144

  float* rin  = alloc(B_ * 4 * D_);
  float* hid  = alloc(B_ * D_);
  float* rbuf = alloc(64);
  float* c0r = alloc((size_t)M2_ * 3 * D_);   float* c0i = alloc((size_t)M2_ * 3 * D_);
  float* qr = alloc(TOK);  float* qi = alloc(TOK);
  float* kr = alloc(TOK);  float* ki = alloc(TOK);
  float* vr = alloc(TOK);  float* vi = alloc(TOK);
  float* simr = alloc((size_t)BH_ * N_ * N_);
  float* simi = alloc((size_t)BH_ * N_ * N_);
  float* aor = alloc(TOK);  float* aoi = alloc(TOK);
  float* m0r = alloc(TOK);  float* m0i = alloc(TOK);
  float* aqr = alloc(TOK);  float* aqi = alloc(TOK);
  float* afr = alloc(TOK);  float* afi = alloc(TOK);
  float* xcr = alloc(TOK);  float* xci = alloc(TOK);
  float* hr  = alloc((size_t)M2_ * HF_);  float* hi = alloc((size_t)M2_ * HF_);
  float* Amr = alloc(DD);  float* Ami = alloc(DD);
  float* ATr = alloc(DD);  float* ATi = alloc(DD);
  float* Ur  = alloc(DD);  float* Ui  = alloc(DD);
  float* P1r = alloc(DD);  float* P1i = alloc(DD);
  float* P2r = alloc(DD);  float* P2i = alloc(DD);
  float* yr  = alloc(TOK); float* yi  = alloc(TOK);

  const long long sQ = (long long)N_ * DH_;   // per-(b,h) q/k/v stride
  const long long sS = (long long)N_ * N_;    // per-(b,h) sim stride

  // ---- router ----
  k_mean_rin<<<4, 256, 0, stream>>>(x_real, x_imag, psi_r, psi_i, rin);
  k_router_hid<<<4, 256, 0, stream>>>(rin, rt_W1, rt_b1, hid);
  k_router_r<<<B_, 256, 0, stream>>>(hid, rt_W2, rt_b2, rbuf);

  // ================= quantum attention (qa) =================
  gemm_c(stream, 0, x_real, x_imag, D_, 0, qa_qkv_Wr, qa_qkv_Wi, D_, 0,
         c0r, c0i, 3 * D_, 0, M2_, 3 * D_, D_, 1, 1.0f, qa_qkv_br, qa_qkv_bi);
  k_split_qkv<<<(unsigned)((3 * TOK + 255) / 256), 256, 0, stream>>>(
      c0r, c0i, qr, qi, kr, ki, vr, vi);
  // sim = SCALE * q conj(k)^T   (batched over 16 (b,h))
  gemm_c(stream, 1, qr, qi, DH_, sQ, kr, ki, DH_, sQ, simr, simi, N_, sS,
         N_, N_, DH_, BH_, 0.125f, nullptr, nullptr);
  k_softmax_abs<<<BH_ * N_, 256, 0, stream>>>(simr, simi);
  // out = attn (real) @ v (complex), B given as [K][N]
  gemm_c(stream, 2, simr, nullptr, N_, sS, vr, vi, DH_, sQ, aor, aoi, DH_, sQ,
         N_, DH_, N_, BH_, 1.0f, nullptr, nullptr);
  k_merge_heads<<<(unsigned)((TOK + 255) / 256), 256, 0, stream>>>(aor, aoi, m0r, m0i);
  gemm_c(stream, 0, m0r, m0i, D_, 0, qa_o_Wr, qa_o_Wi, D_, 0,
         aqr, aqi, D_, 0, M2_, D_, D_, 1, 1.0f, qa_o_br, qa_o_bi);

  // ================= Fourier attention (fa) =================
  gemm_c(stream, 0, x_real, x_imag, D_, 0, fa_qkv_Wr, fa_qkv_Wi, D_, 0,
         c0r, c0i, 3 * D_, 0, M2_, 3 * D_, D_, 1, 1.0f, fa_qkv_br, fa_qkv_bi);
  k_split_qkv<<<(unsigned)((3 * TOK + 255) / 256), 256, 0, stream>>>(
      c0r, c0i, qr, qi, kr, ki, vr, vi);
  k_fft1024<<<BH_ * DH_, 256, 0, stream>>>(qr, qi, -1.0f);
  k_fft1024<<<BH_ * DH_, 256, 0, stream>>>(kr, ki, -1.0f);
  k_fft1024<<<BH_ * DH_, 256, 0, stream>>>(vr, vi, -1.0f);
  gemm_c(stream, 1, qr, qi, DH_, sQ, kr, ki, DH_, sQ, simr, simi, N_, sS,
         N_, N_, DH_, BH_, 1.0f, nullptr, nullptr);
  k_softmax_abs<<<BH_ * N_, 256, 0, stream>>>(simr, simi);
  gemm_c(stream, 2, simr, nullptr, N_, sS, vr, vi, DH_, sQ, aor, aoi, DH_, sQ,
         N_, DH_, N_, BH_, 1.0f, nullptr, nullptr);
  k_fft1024<<<BH_ * DH_, 256, 0, stream>>>(aor, aoi, +1.0f);   // ifft
  k_merge_heads<<<(unsigned)((TOK + 255) / 256), 256, 0, stream>>>(aor, aoi, m0r, m0i);
  gemm_c(stream, 0, m0r, m0i, D_, 0, fa_o_Wr, fa_o_Wi, D_, 0,
         afr, afi, D_, 0, M2_, D_, D_, 1, 1.0f, fa_o_br, fa_o_bi);

  // ---- combine + LN1 ----
  k_combine<<<(unsigned)((TOK + 255) / 256), 256, 0, stream>>>(
      x_real, x_imag, aqr, aqi, afr, afi, rbuf, xcr, xci);
  k_layernorm<<<M2_, 256, 0, stream>>>(xcr, xci, ln1_g, ln1_b);

  // ---- FFN: x + ff2(cgelu(ff1(x))) + LN2 ----
  gemm_c(stream, 0, xcr, xci, D_, 0, ff1_Wr, ff1_Wi, D_, 0,
         hr, hi, HF_, 0, M2_, HF_, D_, 1, 1.0f, ff1_br, ff1_bi);
  const long long HT = (long long)M2_ * HF_;
  k_cgelu<<<(unsigned)((HT + 255) / 256), 256, 0, stream>>>(hr, hi, HT);
  gemm_c(stream, 0, hr, hi, HF_, 0, ff2_Wr, ff2_Wi, HF_, 0,
         yr, yi, D_, 0, M2_, D_, HF_, 1, 1.0f, ff2_br, ff2_bi);
  k_add2<<<(unsigned)((TOK + 255) / 256), 256, 0, stream>>>(xcr, xci, yr, yi, TOK);
  k_layernorm<<<M2_, 256, 0, stream>>>(xcr, xci, ln2_g, ln2_b);

  // ---- Schrodinger: U = expm(-i*dt*H) via Taylor (||A|| ~ 0.02) ----
  k_hbuild<<<(unsigned)((DD + 255) / 256), 256, 0, stream>>>(
      ev_eps_r, ev_eps_i, ev_V_r, ev_V_i, ev_dt, Amr, Ami, ATr, ATi);
  k_uinit<<<(unsigned)((DD + 255) / 256), 256, 0, stream>>>(Amr, Ami, Ur, Ui);
  // P1 = A@A / 2 ; U += P1 ; then P_k = P_{k-1}@A / k
  gemm_c(stream, 0, Amr, Ami, D_, 0, ATr, ATi, D_, 0, P1r, P1i, D_, 0,
         D_, D_, D_, 1, 0.5f, nullptr, nullptr);
  k_add2<<<(unsigned)((DD + 255) / 256), 256, 0, stream>>>(Ur, Ui, P1r, P1i, DD);
  float* Pr = P1r;  float* Pi = P1i;
  float* Qr = P2r;  float* Qi = P2i;
  for (int k = 3; k <= 8; ++k) {
    gemm_c(stream, 0, Pr, Pi, D_, 0, ATr, ATi, D_, 0, Qr, Qi, D_, 0,
           D_, D_, D_, 1, 1.0f / (float)k, nullptr, nullptr);
    k_add2<<<(unsigned)((DD + 255) / 256), 256, 0, stream>>>(Ur, Ui, Qr, Qi, DD);
    float* tr = Pr; Pr = Qr; Qr = tr;
    float* ti = Pi; Pi = Qi; Qi = ti;
  }
  // y = x @ U^T
  gemm_c(stream, 0, xcr, xci, D_, 0, Ur, Ui, D_, 0, yr, yi, D_, 0,
         M2_, D_, D_, 1, 1.0f, nullptr, nullptr);

  // ---- pack output: interleaved complex x, then r ----
  k_pack<<<(unsigned)((TOK + 255) / 256), 256, 0, stream>>>(
      yr, yi, rbuf, (float*)d_out, TOK, (long long)out_size);
}
